// FeatureSteeredConvolutionKerasLayer_49890340111067
// MI455X (gfx1250) — matrix operand order repaired and checked by reading, not compiled, not run on hardware
//
#include <hip/hip_runtime.h>

typedef __attribute__((ext_vector_type(16))) _Float16 v16h;
typedef __attribute__((ext_vector_type(8)))  _Float16 h8;
typedef __attribute__((ext_vector_type(2)))  _Float16 h2;
typedef __attribute__((ext_vector_type(8)))  float    v8f;

#define C_DIM 64
#define D_DIM 64
#define W_DIM 8
#define DEG   16
#define KTOT  512           // W_DIM * C_DIM
#define PITCH 520           // padded row pitch in halves (breaks 1KB-stride bank conflicts)

// ---------------- Kernel 1: x_u = data @ var_u  -> [V, 8] ----------------
__global__ __launch_bounds__(256)
void xu_kernel(const float* __restrict__ data,
               const float* __restrict__ var_u,
               float* __restrict__ xu, int V) {
  int t = blockIdx.x * blockDim.x + threadIdx.x;
  int v = t >> 3;
  int w = t & 7;
  if (v >= V) return;
  const float* row = data + v * C_DIM;
  float s = 0.f;
#pragma unroll
  for (int c = 0; c < C_DIM; ++c)
    s = fmaf(row[c], var_u[c * W_DIM + w], s);
  xu[v * W_DIM + w] = s;
}

// ------------- Kernel 2: fused attention-gather + WMMA GEMM --------------
// 128 threads = 4 waves; each wave owns a 16-vertex tile (64 vertices/block).
__global__ __launch_bounds__(128, 1)
void feast_kernel(const float* __restrict__ data,
                  const int*   __restrict__ edge_dst,
                  const float* __restrict__ edge_vals,
                  const float* __restrict__ xu,
                  const float* __restrict__ var_c,
                  const float* __restrict__ var_w,
                  const float* __restrict__ var_b,
                  float* __restrict__ out, int V) {
  __shared__ _Float16 WT[D_DIM * PITCH];   // var_w transposed: WT[d][k], k = m*64+c
  __shared__ _Float16 Hl[64 * PITCH];      // H tile, f16, [local vertex][k]
  __shared__ float    qbuf[4][2][16][8];   // per-wave softmax weights
  __shared__ int      dstbuf[4][2][16];    // per-wave neighbor indices

  const int tid  = threadIdx.x;
  const int wave = tid >> 5;
  const int lane = tid & 31;
  const int half = (lane >> 4) & 1;        // which 16-lane half of the wave
  const int e    = lane & 15;              // edge slot within vertex
  const int c0   = lane * 2;               // channel pair owned by this lane

  const int waveVBase = blockIdx.x * 64 + wave * 16;

  // ---- stage var_w into LDS, transposed + converted to f16 ----
  for (int i = tid; i < D_DIM * KTOT; i += 128) {
    int d = i >> 9;         // i / 512
    int k = i & 511;        // i % 512
    WT[d * PITCH + k] = (_Float16)var_w[k * D_DIM + d];
  }

  float vc[8];
#pragma unroll
  for (int m = 0; m < 8; ++m) vc[m] = var_c[m];

  // =============== Phase A: per-edge softmax + neighbor aggregation ======
  for (int p = 0; p < 8; ++p) {
    __syncthreads();                       // WAR on qbuf/dstbuf; covers WT fill on p==0
    int v = waveVBase + 2 * p + half;
    if (v < V) {
      int   eidx = v * DEG + e;
      int   dst  = edge_dst[eidx];
      float ev   = edge_vals[eidx];
      float4 xs0 = *(const float4*)(xu + v * 8);
      float4 xs1 = *(const float4*)(xu + v * 8 + 4);
      float4 xd0 = *(const float4*)(xu + dst * 8);
      float4 xd1 = *(const float4*)(xu + dst * 8 + 4);
      float q[8];
      q[0] = xs0.x - xd0.x + vc[0];  q[1] = xs0.y - xd0.y + vc[1];
      q[2] = xs0.z - xd0.z + vc[2];  q[3] = xs0.w - xd0.w + vc[3];
      q[4] = xs1.x - xd1.x + vc[4];  q[5] = xs1.y - xd1.y + vc[5];
      q[6] = xs1.z - xd1.z + vc[6];  q[7] = xs1.w - xd1.w + vc[7];
      float mx = q[0];
#pragma unroll
      for (int m = 1; m < 8; ++m) mx = fmaxf(mx, q[m]);
      float s = 0.f;
#pragma unroll
      for (int m = 0; m < 8; ++m) { q[m] = __expf(q[m] - mx); s += q[m]; }
      float inv = ev / s;                  // fold edge_vals (1/DEG) into q
#pragma unroll
      for (int m = 0; m < 8; ++m) qbuf[wave][half][e][m] = q[m] * inv;
      dstbuf[wave][half][e] = dst;
    } else {
      dstbuf[wave][half][e] = 0;
#pragma unroll
      for (int m = 0; m < 8; ++m) qbuf[wave][half][e][m] = 0.f;
    }
    __syncthreads();

#pragma unroll
    for (int vv = 0; vv < 2; ++vv) {
      int vA = waveVBase + 2 * p + vv;
      if (vA >= V) continue;
      float acc[16];
#pragma unroll
      for (int i = 0; i < 16; ++i) acc[i] = 0.f;
#pragma unroll
      for (int ee = 0; ee < 16; ++ee) {
        int dst = dstbuf[wave][vv][ee];                       // LDS broadcast
        float2 dv = *(const float2*)(data + dst * C_DIM + c0); // coalesced 256B/row
        const float4* qp = (const float4*)qbuf[wave][vv][ee];
        float4 q0 = qp[0], q1 = qp[1];
        acc[0]  = fmaf(q0.x, dv.x, acc[0]);  acc[1]  = fmaf(q0.x, dv.y, acc[1]);
        acc[2]  = fmaf(q0.y, dv.x, acc[2]);  acc[3]  = fmaf(q0.y, dv.y, acc[3]);
        acc[4]  = fmaf(q0.z, dv.x, acc[4]);  acc[5]  = fmaf(q0.z, dv.y, acc[5]);
        acc[6]  = fmaf(q0.w, dv.x, acc[6]);  acc[7]  = fmaf(q0.w, dv.y, acc[7]);
        acc[8]  = fmaf(q1.x, dv.x, acc[8]);  acc[9]  = fmaf(q1.x, dv.y, acc[9]);
        acc[10] = fmaf(q1.y, dv.x, acc[10]); acc[11] = fmaf(q1.y, dv.y, acc[11]);
        acc[12] = fmaf(q1.z, dv.x, acc[12]); acc[13] = fmaf(q1.z, dv.y, acc[13]);
        acc[14] = fmaf(q1.w, dv.x, acc[14]); acc[15] = fmaf(q1.w, dv.y, acc[15]);
      }
      int lvert = wave * 16 + 2 * p + vv;
#pragma unroll
      for (int m = 0; m < 8; ++m) {
        h2 hv;
        hv[0] = (_Float16)acc[2 * m];
        hv[1] = (_Float16)acc[2 * m + 1];
        *(h2*)(&Hl[lvert * PITCH + m * C_DIM + c0]) = hv;     // k = m*64 + {2l,2l+1}
      }
    }
  }
  __syncthreads();

  // =============== Phase B: y_tile = H_tile(16x512) @ W(512x64) via WMMA =
  // A frag (16x32 f16): lane m-row = lane&15; halves 0..15 = contiguous K
  //                     from kb = kc*32 + half*16.
  // B frag (32x16 f16): lane n-col = lane&15; halves 0..15 = contiguous K
  //                     from kc*32 + half*16 in the transposed WT rows.
  v8f cacc[4] = {};
  const int mrow = lane & 15;
  const _Float16* Hrow = &Hl[(wave * 16 + mrow) * PITCH];

  for (int kc = 0; kc < 16; ++kc) {
    union { v16h v; h8 p[2]; } a;
    int kbA = kc * 32 + half * 16;
    a.p[0] = *(const h8*)(Hrow + kbA);        // ds_load_b128
    a.p[1] = *(const h8*)(Hrow + kbA + 8);    // ds_load_b128
    int kbB = kc * 32 + half * 16;
#pragma unroll
    for (int n = 0; n < 4; ++n) {
      union { v16h v; h8 p[2]; } b;
      const _Float16* Wrow = &WT[(n * 16 + mrow) * PITCH];
      b.p[0] = *(const h8*)(Wrow + kbB);
      b.p[1] = *(const h8*)(Wrow + kbB + 8);
      cacc[n] = __builtin_amdgcn_wmma_f32_16x16x32_f16(
          /*neg_a=*/false, a.v, /*neg_b=*/false, b.v,
          /*c_mod=*/(short)0, cacc[n], /*reuse_a=*/false, /*reuse_b=*/false);
    }
  }

  // ---- bias + store: D layout -> lane col = lane&15, VGPR r -> row r + half*8
#pragma unroll
  for (int n = 0; n < 4; ++n) {
    int col = n * 16 + mrow;
    float bias = var_b[col];
#pragma unroll
    for (int r = 0; r < 8; ++r) {
      int vOut = waveVBase + r + half * 8;
      if (vOut < V) out[vOut * D_DIM + col] = cacc[n][r] + bias;
    }
  }
}

extern "C" void kernel_launch(void* const* d_in, const int* in_sizes, int n_in,
                              void* d_out, int out_size, void* d_ws, size_t ws_size,
                              hipStream_t stream) {
  const float* data      = (const float*)d_in[0];
  // d_in[1] = edge_src: sorted repeat(arange(V), DEG) -> implicit, unused
  const int*   edge_dst  = (const int*)  d_in[2];
  const float* edge_vals = (const float*)d_in[3];
  const float* var_u     = (const float*)d_in[4];
  const float* var_c     = (const float*)d_in[5];
  const float* var_w     = (const float*)d_in[6];
  const float* var_b     = (const float*)d_in[7];
  float* out = (float*)d_out;

  int V = in_sizes[0] / C_DIM;
  float* xu = (float*)d_ws;                // V*8 floats = 3.2 MB scratch

  dim3 b1(256), g1((V * W_DIM + 255) / 256);
  xu_kernel<<<g1, b1, 0, stream>>>(data, var_u, xu, V);

  dim3 b2(128), g2((V + 63) / 64);
  feast_kernel<<<g2, b2, 0, stream>>>(data, edge_dst, edge_vals, xu,
                                      var_c, var_w, var_b, out, V);
}
